// LinAttnBlock_53575422050940
// MI455X (gfx1250) — compile-verified, matrix operand order checked by reading, not run on hardware
//
#include <hip/hip_runtime.h>
#include <hip/hip_bf16.h>

// ---------------------------------------------------------------------------
// LinAttnBlock for MI455X (gfx1250): bf16 WMMA pipeline with LDS-staged,
// double-buffered GEMM core (async global->LDS copies, ASYNCcnt tracked).
//   xT   = transpose(x) as bf16                     [b][l][c]
//   QKV  = W_qkv(bf16) x xT    (v_wmma_f32_16x16x32_bf16, fp32 acc)
//          -> qT bf16 [b][l][d] (packed stores), k fp32 [b][d][l], v bf16 [b][e][n]
//   ks   = softmax_l(k) bf16   [b][d][n]
//   ctx  = ks x v^T            [b][d][e]  (WMMA)
//   W2   = W_out x ctx^T       [b][o][d]  (WMMA)  -- algebraic fusion
//   out  = W2 x q + b_out      [b][o][l]  (WMMA, fp32 out)
// ---------------------------------------------------------------------------

typedef __attribute__((ext_vector_type(16))) __bf16 v16bf;
typedef __attribute__((ext_vector_type(8)))  float  v8f;

#define B_  16
#define C_  512
#define L_  4096
#define M3_ 1536

#define TILE_M 128
#define TILE_N 128
#define TILE_K 32

#if defined(__AMDGCN__) && __has_builtin(__builtin_amdgcn_global_load_async_to_lds_b128)
#define USE_ASYNC 1
#else
#define USE_ASYNC 0
#endif

union FragU { v16bf v; uint4 q[2]; };

__device__ __forceinline__ __bf16 f2bf(float f) {
  unsigned u = __float_as_uint(f);
  u = (u + 0x7FFFu + ((u >> 16) & 1u)) >> 16;   // round-to-nearest-even
  unsigned short s = (unsigned short)u;
  __bf16 r;
  __builtin_memcpy(&r, &s, 2);
  return r;
}

__device__ __forceinline__ v8f wmma_bf16(v16bf a, v16bf b, v8f c) {
  return __builtin_amdgcn_wmma_f32_16x16x32_bf16(
      /*neg_a=*/false, a, /*neg_b=*/false, b,
      /*c_mod=*/(short)0, c, /*reuse_a=*/false, /*reuse_b=*/false);
}

#if USE_ASYNC
// Builtin signature (from compiler diagnostic): operands are pointers to
// int __attribute__((vector_size(16))) in global (AS1) / LDS (AS3) spaces.
typedef int v4i_vs __attribute__((vector_size(16)));
typedef __attribute__((address_space(1))) v4i_vs* gptr_v4i;
typedef __attribute__((address_space(3))) v4i_vs* lptr_v4i;
#endif

// 16-byte global -> LDS copy: async (ASYNCcnt) on gfx1250, sync fallback.
__device__ __forceinline__ void copy16_g2l(const __bf16* gp, __bf16* lp) {
#if USE_ASYNC
  __builtin_amdgcn_global_load_async_to_lds_b128(
      (gptr_v4i)(void*)gp, (lptr_v4i)(void*)lp, /*offset=*/0, /*cpol=*/0);
#else
  *(uint4*)lp = *(const uint4*)gp;
#endif
}

__device__ __forceinline__ void async_wait0() {
#if USE_ASYNC
#if __has_builtin(__builtin_amdgcn_s_wait_asynccnt)
  __builtin_amdgcn_s_wait_asynccnt(0);
#else
  asm volatile("s_wait_asynccnt 0x0" ::: "memory");
#endif
#endif
}

// Stage a TILE_M x TILE_K bf16 tile (row-major, K contiguous) into LDS.
// 256 threads x 2 chunks of 16B each = 8 KB.
__device__ __forceinline__ void stage_tile(const __bf16* __restrict__ src, int ld,
                                           int rowbase, int k,
                                           __bf16 (*dst)[TILE_K]) {
  const int t = threadIdx.x;
#pragma unroll
  for (int h = 0; h < 2; ++h) {
    const int c   = t + h * 256;        // 0..511
    const int row = c >> 2;             // 0..127
    const int col = (c & 3) * 8;        // 0,8,16,24
    copy16_g2l(src + (size_t)(rowbase + row) * ld + (k + col), &dst[row][col]);
  }
}

// 16-bit A/B fragment from an LDS tile (CDNA5 §7.12.2 layout):
// lane r=lane&15 owns one row; lanes 0-15 K-chunks {0..7,16..23},
// lanes 16-31 K-chunks {8..15,24..31}. Two ds_load_b128 per fragment.
__device__ __forceinline__ v16bf lds_frag(const __bf16 (*tile)[TILE_K], int rbase) {
  const int lane = threadIdx.x & 31;
  const int r    = rbase + (lane & 15);
  const int k0   = (lane & 16) ? 8 : 0;
  FragU f;
  f.q[0] = *(const uint4*)&tile[r][k0];
  f.q[1] = *(const uint4*)&tile[r][k0 + 16];
  return f.v;
}

// Double-buffered 128x128 macro-tile GEMM core.
// D[m,n] = sum_k A[m][k] * B[n][k]  (both operands row-major with K contiguous)
// 8 waves: wave computes 32(M) x 64(N) -> 8 WMMAs per 32-wide k-step.
__device__ __forceinline__ void gemm_core(
    const __bf16* __restrict__ A,  int lda, int mblk,
    const __bf16* __restrict__ Bn, int ldb, int nblk, int K,
    __bf16 (*At)[TILE_M][TILE_K], __bf16 (*Bt)[TILE_M][TILE_K],
    v8f (&acc)[2][4]) {
  const int wave = threadIdx.x >> 5;
  const int wm   = (wave >> 1) * 32;   // 0,32,64,96
  const int wn   = (wave & 1) * 64;    // 0,64
  stage_tile(A,  lda, mblk, 0, At[0]);
  stage_tile(Bn, ldb, nblk, 0, Bt[0]);
  const int nk = K / TILE_K;
  int buf = 0;
  for (int kt = 0; kt < nk; ++kt) {
    async_wait0();          // our wave's stage of `buf` is complete
    __syncthreads();        // everyone's stage complete; prior reads retired
    if (kt + 1 < nk) {      // prefetch next k-slice into the other buffer
      stage_tile(A,  lda, mblk, (kt + 1) * TILE_K, At[buf ^ 1]);
      stage_tile(Bn, ldb, nblk, (kt + 1) * TILE_K, Bt[buf ^ 1]);
    }
    v16bf a0 = lds_frag(At[buf], wm);
    v16bf a1 = lds_frag(At[buf], wm + 16);
#pragma unroll
    for (int j = 0; j < 4; ++j) {
      v16bf b = lds_frag(Bt[buf], wn + 16 * j);
      acc[0][j] = wmma_bf16(a0, b, acc[0][j]);
      acc[1][j] = wmma_bf16(a1, b, acc[1][j]);
    }
    buf ^= 1;
  }
}

// --------------------------- elementwise convert ---------------------------
__global__ __launch_bounds__(256) void k_f32_to_bf16(const float* __restrict__ src,
                                                     __bf16* __restrict__ dst, int n) {
  int i = blockIdx.x * blockDim.x + threadIdx.x;
  if (i < n) dst[i] = f2bf(src[i]);
}

// --------------------- x [b][c][l] f32 -> xT [b][l][c] bf16 ----------------
__global__ __launch_bounds__(256) void k_transpose_x(const float* __restrict__ x,
                                                     __bf16* __restrict__ xT) {
  __shared__ __bf16 tile[32][33];
  const int b  = blockIdx.z;
  const int c0 = blockIdx.y * 32;
  const int l0 = blockIdx.x * 32;
  const float* xb  = x  + (size_t)b * C_ * L_;
  __bf16*      xTb = xT + (size_t)b * L_ * C_;
  for (int i = threadIdx.y; i < 32; i += 8)
    tile[i][threadIdx.x] = f2bf(xb[(size_t)(c0 + i) * L_ + (l0 + threadIdx.x)]);
  __syncthreads();
  for (int i = threadIdx.y; i < 32; i += 8)
    xTb[(size_t)(l0 + i) * C_ + (c0 + threadIdx.x)] = tile[threadIdx.x][i];
}

// ------------------------------- QKV GEMM ----------------------------------
// grid (L/128, 1536/128, B), block 256 (8 waves)
__global__ __launch_bounds__(256) void k_qkv_gemm(const __bf16* __restrict__ Wq,
                                                  const __bf16* __restrict__ xT,
                                                  __bf16* __restrict__ qT,
                                                  float*  __restrict__ kf,
                                                  __bf16* __restrict__ v) {
  __shared__ __bf16 At[2][TILE_M][TILE_K];
  __shared__ __bf16 Bt[2][TILE_M][TILE_K];
  const int b    = blockIdx.z;
  const int mblk = blockIdx.y * TILE_M;            // o in [0,1536)
  const int nblk = blockIdx.x * TILE_N;            // l
  const __bf16* Xb = xT + (size_t)b * L_ * C_;     // [l][c], ld=C
  v8f acc[2][4] = {};
  gemm_core(Wq, C_, mblk, Xb, C_, nblk, C_, At, Bt, acc);

  const int wave = threadIdx.x >> 5, lane = threadIdx.x & 31;
  const int wm = (wave >> 1) * 32, wn = (wave & 1) * 64;
  const int nloc = lane & 15, mo = (lane & 16) ? 8 : 0;
  if (mblk < C_) {                        // Q -> qT[b][l][d]: packed b128 stores
    __bf16* qb = qT + (size_t)b * L_ * C_;
#pragma unroll
    for (int sm = 0; sm < 2; ++sm)
#pragma unroll
      for (int j = 0; j < 4; ++j) {
        const int l = nblk + wn + 16 * j + nloc;
        union { __bf16 h[8]; uint4 q; } u;
#pragma unroll
        for (int g = 0; g < 8; ++g) u.h[g] = f2bf(acc[sm][j][g]);
        *(uint4*)(qb + (size_t)l * C_ + (mblk + wm + sm * 16 + mo)) = u.q;
      }
  } else if (mblk < 2 * C_) {             // K -> kf[b][d][l] fp32 (for softmax)
    float* kb = kf + (size_t)b * C_ * L_;
#pragma unroll
    for (int sm = 0; sm < 2; ++sm)
#pragma unroll
      for (int j = 0; j < 4; ++j) {
        const int l = nblk + wn + 16 * j + nloc;
        const int d0 = mblk - C_ + wm + sm * 16 + mo;
#pragma unroll
        for (int g = 0; g < 8; ++g)
          kb[(size_t)(d0 + g) * L_ + l] = acc[sm][j][g];
      }
  } else {                                // V -> v[b][e][n] bf16 row-major
    __bf16* vb = v + (size_t)b * C_ * L_;
#pragma unroll
    for (int sm = 0; sm < 2; ++sm)
#pragma unroll
      for (int j = 0; j < 4; ++j) {
        const int n = nblk + wn + 16 * j + nloc;
        const int e0 = mblk - 2 * C_ + wm + sm * 16 + mo;
#pragma unroll
        for (int g = 0; g < 8; ++g)
          vb[(size_t)(e0 + g) * L_ + n] = f2bf(acc[sm][j][g]);
      }
  }
}

// -------------------- softmax over l, one block per (b,d) row --------------
__global__ __launch_bounds__(256) void k_softmax(const float* __restrict__ kf,
                                                 __bf16* __restrict__ ks) {
  const int row = blockIdx.x;                      // b*C + d
  const float* src = kf + (size_t)row * L_;
  __bf16*      dst = ks + (size_t)row * L_;
  __shared__ float red[256];
  float m = -INFINITY;
  for (int i = threadIdx.x; i < L_; i += 256) m = fmaxf(m, src[i]);
  red[threadIdx.x] = m; __syncthreads();
  for (int s = 128; s > 0; s >>= 1) {
    if (threadIdx.x < s) red[threadIdx.x] = fmaxf(red[threadIdx.x], red[threadIdx.x + s]);
    __syncthreads();
  }
  m = red[0]; __syncthreads();
  float acc = 0.f;
  for (int i = threadIdx.x; i < L_; i += 256) acc += __expf(src[i] - m);
  red[threadIdx.x] = acc; __syncthreads();
  for (int s = 128; s > 0; s >>= 1) {
    if (threadIdx.x < s) red[threadIdx.x] += red[threadIdx.x + s];
    __syncthreads();
  }
  const float inv = 1.0f / red[0];
  for (int i = threadIdx.x; i < L_; i += 256)
    dst[i] = f2bf(__expf(src[i] - m) * inv);
}

// ------------------- ctx[b][d][e] = sum_n ks[d][n] * v[e][n] ---------------
__global__ __launch_bounds__(256) void k_ctx_gemm(const __bf16* __restrict__ ks,
                                                  const __bf16* __restrict__ v,
                                                  __bf16* __restrict__ ctx) {
  __shared__ __bf16 At[2][TILE_M][TILE_K];
  __shared__ __bf16 Bt[2][TILE_M][TILE_K];
  const int b    = blockIdx.z;
  const int mblk = blockIdx.y * TILE_M;            // d
  const int nblk = blockIdx.x * TILE_N;            // e
  v8f acc[2][4] = {};
  gemm_core(ks + (size_t)b * C_ * L_, L_, mblk,
            v  + (size_t)b * C_ * L_, L_, nblk, L_, At, Bt, acc);
  __bf16* cb = ctx + (size_t)b * C_ * C_;
  const int wave = threadIdx.x >> 5, lane = threadIdx.x & 31;
  const int wm = (wave >> 1) * 32, wn = (wave & 1) * 64;
  const int nloc = lane & 15, mo = (lane & 16) ? 8 : 0;
#pragma unroll
  for (int sm = 0; sm < 2; ++sm)
#pragma unroll
    for (int j = 0; j < 4; ++j) {
      const int e  = nblk + wn + 16 * j + nloc;
      const int d0 = mblk + wm + sm * 16 + mo;
#pragma unroll
      for (int g = 0; g < 8; ++g)
        cb[(size_t)(d0 + g) * C_ + e] = f2bf(acc[sm][j][g]);
    }
}

// --------------- W2[b][o][d] = sum_e w_out[o][e] * ctx[b][d][e] ------------
__global__ __launch_bounds__(256) void k_w2_gemm(const __bf16* __restrict__ wout,
                                                 const __bf16* __restrict__ ctx,
                                                 __bf16* __restrict__ W2) {
  __shared__ __bf16 At[2][TILE_M][TILE_K];
  __shared__ __bf16 Bt[2][TILE_M][TILE_K];
  const int b    = blockIdx.z;
  const int mblk = blockIdx.y * TILE_M;            // o
  const int nblk = blockIdx.x * TILE_N;            // d
  v8f acc[2][4] = {};
  gemm_core(wout, C_, mblk,
            ctx + (size_t)b * C_ * C_, C_, nblk, C_, At, Bt, acc);
  __bf16* wb = W2 + (size_t)b * C_ * C_;
  const int wave = threadIdx.x >> 5, lane = threadIdx.x & 31;
  const int wm = (wave >> 1) * 32, wn = (wave & 1) * 64;
  const int nloc = lane & 15, mo = (lane & 16) ? 8 : 0;
#pragma unroll
  for (int sm = 0; sm < 2; ++sm)
#pragma unroll
    for (int j = 0; j < 4; ++j) {
      const int d  = nblk + wn + 16 * j + nloc;
      const int o0 = mblk + wm + sm * 16 + mo;
#pragma unroll
      for (int g = 0; g < 8; ++g)
        wb[(size_t)(o0 + g) * C_ + d] = f2bf(acc[sm][j][g]);
    }
}

// ---------- out[b][o][l] = sum_d W2[b][o][d] * qT[b][l][d] + b_out[o] ------
__global__ __launch_bounds__(256) void k_final_gemm(const __bf16* __restrict__ W2,
                                                    const __bf16* __restrict__ qT,
                                                    const float* __restrict__ b_out,
                                                    float* __restrict__ out) {
  __shared__ __bf16 At[2][TILE_M][TILE_K];
  __shared__ __bf16 Bt[2][TILE_M][TILE_K];
  const int b    = blockIdx.z;
  const int mblk = blockIdx.y * TILE_M;            // o
  const int nblk = blockIdx.x * TILE_N;            // l
  v8f acc[2][4] = {};
  gemm_core(W2 + (size_t)b * C_ * C_, C_, mblk,
            qT + (size_t)b * L_ * C_, C_, nblk, C_, At, Bt, acc);
  float* ob = out + (size_t)b * C_ * L_;
  const int wave = threadIdx.x >> 5, lane = threadIdx.x & 31;
  const int wm = (wave >> 1) * 32, wn = (wave & 1) * 64;
  const int nloc = lane & 15, mo = (lane & 16) ? 8 : 0;
#pragma unroll
  for (int sm = 0; sm < 2; ++sm)
#pragma unroll
    for (int j = 0; j < 4; ++j) {
      const int l  = nblk + wn + 16 * j + nloc;
      const int o0 = mblk + wm + sm * 16 + mo;
#pragma unroll
      for (int g = 0; g < 8; ++g)
        ob[(size_t)(o0 + g) * L_ + l] = acc[sm][j][g] + b_out[o0 + g];
    }
}

// ---------------------------------------------------------------------------
extern "C" void kernel_launch(void* const* d_in, const int* in_sizes, int n_in,
                              void* d_out, int out_size, void* d_ws, size_t ws_size,
                              hipStream_t stream) {
  const float* x     = (const float*)d_in[0];   // [16][512][4096]
  const float* w_qkv = (const float*)d_in[1];   // [1536][512]
  const float* w_out = (const float*)d_in[2];   // [512][512]
  const float* b_out = (const float*)d_in[3];   // [512]
  float* out = (float*)d_out;                   // [16][512][4096]

  // workspace carve-up (~421 MB total)
  char* ws = (char*)d_ws;
  size_t off = 0;
  auto carve = [&](size_t bytes) {
    char* p = ws + off;
    off += (bytes + 255) & ~(size_t)255;
    return p;
  };
  __bf16* xT   = (__bf16*)carve((size_t)B_ * L_ * C_ * 2);   // 64 MB
  __bf16* qT   = (__bf16*)carve((size_t)B_ * L_ * C_ * 2);   // 64 MB
  float*  kf   = (float*) carve((size_t)B_ * C_ * L_ * 4);   // 128 MB
  __bf16* ksft = (__bf16*)carve((size_t)B_ * C_ * L_ * 2);   // 64 MB
  __bf16* vbuf = (__bf16*)carve((size_t)B_ * C_ * L_ * 2);   // 64 MB
  __bf16* ctx  = (__bf16*)carve((size_t)B_ * C_ * C_ * 2);   // 8 MB
  __bf16* W2   = (__bf16*)carve((size_t)B_ * C_ * C_ * 2);   // 8 MB
  __bf16* wqb  = (__bf16*)carve((size_t)M3_ * C_ * 2);
  __bf16* wob  = (__bf16*)carve((size_t)C_ * C_ * 2);

  // 1) convert weights to bf16
  k_f32_to_bf16<<<(M3_ * C_ + 255) / 256, 256, 0, stream>>>(w_qkv, wqb, M3_ * C_);
  k_f32_to_bf16<<<(C_ * C_ + 255) / 256, 256, 0, stream>>>(w_out, wob, C_ * C_);

  // 2) transpose-convert x -> xT bf16 [b][l][c]
  k_transpose_x<<<dim3(L_ / 32, C_ / 32, B_), dim3(32, 8), 0, stream>>>(x, xT);

  // 3) QKV GEMM (bf16 WMMA, LDS double-buffered, async staging)
  k_qkv_gemm<<<dim3(L_ / TILE_N, M3_ / TILE_M, B_), 256, 0, stream>>>(wqb, xT, qT, kf, vbuf);

  // 4) softmax over sequence dim for K
  k_softmax<<<B_ * C_, 256, 0, stream>>>(kf, ksft);

  // 5) context = K_soft x V^T
  k_ctx_gemm<<<dim3(C_ / TILE_N, C_ / TILE_M, B_), 256, 0, stream>>>(ksft, vbuf, ctx);

  // 6) W2 = W_out x ctx^T (fusion: removes a [512x4096] GEMM + 128MB traffic)
  k_w2_gemm<<<dim3(C_ / TILE_N, C_ / TILE_M, B_), 256, 0, stream>>>(wob, ctx, W2);

  // 7) out = W2 x Q + b_out
  k_final_gemm<<<dim3(L_ / TILE_N, C_ / TILE_M, B_), 256, 0, stream>>>(W2, qT, b_out, out);
}
